// GDN_55757265436873
// MI455X (gfx1250) — matrix operand order repaired and checked by reading, not compiled, not run on hardware
//
#include <hip/hip_runtime.h>
#include <hip/hip_bf16.h>

typedef __attribute__((ext_vector_type(16))) _Float16 v16h;
typedef __attribute__((ext_vector_type(8)))  _Float16 v8h;
typedef __attribute__((ext_vector_type(8)))  float    v8f;
typedef __attribute__((ext_vector_type(4)))  float    v4f;

constexpr int NN = 20000;   // sensors
constexpr int WW = 64;      // window
constexpr int DD = 64;      // embed dim
constexpr int HH = 64;      // hidden
constexpr int EE = 640000;  // edges
constexpr int BB = 2;       // batch
constexpr int NT = NN / 16; // 1250 row tiles (exact)

// ---------------------------------------------------------------- WMMA helpers
__device__ inline v8f wmma_f16(v16h a, v16h b, v8f c) {
  // v_wmma_f32_16x16x32_f16  (probe-confirmed 8-arg signature)
  return __builtin_amdgcn_wmma_f32_16x16x32_f16(false, a, false, b, (short)0, c,
                                                false, false);
}

// A-matrix 16x32 f16 layout (ISA 7.12.2): lane(<16)=row m, hh=lane>>4.
// Lane's 16 elements = rows m, K in two contiguous runs:
//   [kb + hh*8, +8)  and  [kb + 16 + hh*8, +8)
// -> 4x b128 global loads (f32) or 2x b128 LDS loads (f16).
__device__ inline v16h load_A_g(const float* base, int ldk, int m, int hh, int kb) {
  const float* r = base + (size_t)m * ldk + kb + hh * 8;
  v4f p0 = *(const v4f*)(r + 0);
  v4f p1 = *(const v4f*)(r + 4);
  v4f p2 = *(const v4f*)(r + 16);
  v4f p3 = *(const v4f*)(r + 20);
  v16h a;
#pragma unroll
  for (int i = 0; i < 4; ++i) {
    a[i]      = (_Float16)p0[i];
    a[4 + i]  = (_Float16)p1[i];
    a[8 + i]  = (_Float16)p2[i];
    a[12 + i] = (_Float16)p3[i];
  }
  return a;
}

__device__ inline v16h load_A_l(const _Float16* base, int ldk, int m, int hh, int kb) {
  const _Float16* r = base + m * ldk + kb + hh * 8;
  v8h lo = *(const v8h*)(r);        // K run 0..7
  v8h hi = *(const v8h*)(r + 16);   // K run 16..23
  return __builtin_shufflevector(lo, hi, 0, 1, 2, 3, 4, 5, 6, 7,
                                 8, 9, 10, 11, 12, 13, 14, 15);
}

// B-matrix 32x16 f16 (KxN): lane col n, hh=lane>>4; element i of lane's
// fragment = B[kb + hh*8 + (i&7) + (i>>3)*16][n] (same K order as A).
// Fragments are pre-swizzled into LDS, one contiguous v16h per lane.
__device__ inline v16h make_B_frag(const float* W, int ldn, int kc, int ct, int ln) {
  int fhh = ln >> 4, fn = ln & 15;
  v16h b;
#pragma unroll
  for (int i = 0; i < 16; ++i)
    b[i] = (_Float16)W[(size_t)(kc * 32 + fhh * 8 + (i & 7) + (i >> 3) * 16) * ldn + ct * 16 + fn];
  return b;
}

// ---------------------------------------------------------------- init
__global__ void gdn_init(float* seg_max, float* seg_sum, float* agg) {
  int i = blockIdx.x * blockDim.x + threadIdx.x;
  if (i < NN) { seg_max[i] = -__builtin_huge_valf(); seg_sum[i] = 0.f; }
  if (i < BB * NN * WW) agg[i] = 0.f;
}

// ---------------------------------------------------------------- q/k GEMMs
__global__ void gdn_qk(const float* __restrict__ emb, const float* __restrict__ Wq,
                       const float* __restrict__ Wk, float* __restrict__ q,
                       float* __restrict__ k) {
  __shared__ v16h sWqF[2 * 4 * 32];  // [kc][ct][lane] fragments, 8 KB
  __shared__ v16h sWkF[2 * 4 * 32];

  for (int f = threadIdx.x; f < 2 * 4 * 32; f += blockDim.x) {
    int kc = f >> 7, ct = (f >> 5) & 3, ln = f & 31;
    sWqF[f] = make_B_frag(Wq, DD, kc, ct, ln);
    sWkF[f] = make_B_frag(Wk, DD, kc, ct, ln);
  }
  __syncthreads();

  int wave = threadIdx.x >> 5, lane = threadIdx.x & 31;
  int tile = blockIdx.x * 8 + wave;
  if (tile >= NT) return;
  int m = lane & 15, hh = lane >> 4, n = lane & 15;
  int row0 = tile * 16;

  v16h a0 = load_A_g(emb + (size_t)row0 * DD, DD, m, hh, 0);
  v16h a1 = load_A_g(emb + (size_t)row0 * DD, DD, m, hh, 32);

#pragma unroll
  for (int ct = 0; ct < 4; ++ct) {
    v8f accq = {}, acck = {};
    accq = wmma_f16(a0, sWqF[(0 * 4 + ct) * 32 + lane], accq);
    accq = wmma_f16(a1, sWqF[(1 * 4 + ct) * 32 + lane], accq);
    acck = wmma_f16(a0, sWkF[(0 * 4 + ct) * 32 + lane], acck);
    acck = wmma_f16(a1, sWkF[(1 * 4 + ct) * 32 + lane], acck);
#pragma unroll
    for (int r = 0; r < 8; ++r) {
      size_t o = (size_t)(row0 + r + 8 * hh) * DD + ct * 16 + n;
      q[o] = accq[r];
      k[o] = acck[r];
    }
  }
}

// ---------------------------------------------------------------- edge scores
__global__ void gdn_score(const int* __restrict__ ei, const float* __restrict__ q,
                          const float* __restrict__ k, const float* __restrict__ v,
                          float* __restrict__ score) {
  int e = blockIdx.x * 8 + (threadIdx.x >> 5);
  int lane = threadIdx.x & 31;
  if (e >= EE) return;
  int s = ei[e], d = ei[EE + e];
  float p = 0.f;
#pragma unroll
  for (int j = 0; j < 2; ++j) {
    int dd = lane + 32 * j;
    float val = q[(size_t)s * DD + dd] + k[(size_t)d * DD + dd];
    p += tanhf(val) * v[dd];
  }
#pragma unroll
  for (int off = 16; off > 0; off >>= 1) p += __shfl_xor(p, off, 32);
  if (lane == 0) score[e] = p;
}

// ---------------------------------------------------------------- segment max
__device__ inline void atomicMaxF(float* addr, float val) {
  if (val >= 0.f) atomicMax((int*)addr, __float_as_int(val));
  else            atomicMin((unsigned int*)addr, __float_as_uint(val));
}

__global__ void gdn_segmax(const int* __restrict__ ei,
                           const float* __restrict__ score,
                           float* __restrict__ seg_max) {
  int e = blockIdx.x * blockDim.x + threadIdx.x;
  if (e >= EE) return;
  atomicMaxF(&seg_max[ei[EE + e]], score[e]);
}

// ---------------------------------------------------------------- exp + sum
__global__ void gdn_expsum(const int* __restrict__ ei,
                           const float* __restrict__ seg_max,
                           float* __restrict__ score,  // in: score, out: exp
                           float* __restrict__ seg_sum) {
  int e = blockIdx.x * blockDim.x + threadIdx.x;
  if (e >= EE) return;
  int d = ei[EE + e];
  float ex = expf(score[e] - seg_max[d]);
  score[e] = ex;
  atomicAdd(&seg_sum[d], ex);
}

// ---------------------------------------------------------------- aggregation
__global__ void gdn_agg(const int* __restrict__ ei, const float* __restrict__ x,
                        const float* __restrict__ exps,
                        const float* __restrict__ seg_sum,
                        float* __restrict__ agg) {
  int e = blockIdx.x * 8 + (threadIdx.x >> 5);
  int lane = threadIdx.x & 31;
  if (e >= EE) return;
  int s = ei[e], d = ei[EE + e];
  float attn = exps[e] / (seg_sum[d] + 1e-8f);
#pragma unroll
  for (int b = 0; b < BB; ++b) {
#pragma unroll
    for (int j = 0; j < 2; ++j) {
      int w = lane + 32 * j;
      float val = attn * x[((size_t)b * NN + s) * WW + w];
      atomicAdd(&agg[((size_t)b * NN + d) * WW + w], val);
    }
  }
}

// ---------------------------------------------------------------- fused MLP
__global__ void gdn_mlp(const float* __restrict__ x, const float* __restrict__ agg,
                        const float* __restrict__ fcW, const float* __restrict__ fcb,
                        const float* __restrict__ f1W, const float* __restrict__ f1b,
                        const float* __restrict__ f2W, const float* __restrict__ f2b,
                        float* __restrict__ pred) {
  __shared__ v16h sFcF[4 * 4 * 32];          // fcW fragments [kc][ct][lane], 16 KB
  __shared__ v16h sF1F[2 * 2 * 32];          // f1W fragments, 4 KB
  __shared__ float sFcb[HH], sF1b[HH / 2], sF2[HH / 2];
  __shared__ _Float16 sH[8][16 * HH];        // per-wave h tile (row-major), 16 KB
  __shared__ float sMid[8][16 * (HH / 2)];   // per-wave mid products, 16 KB

  for (int f = threadIdx.x; f < 4 * 4 * 32; f += blockDim.x) {
    int kc = f >> 7, ct = (f >> 5) & 3, ln = f & 31;
    sFcF[f] = make_B_frag(fcW, HH, kc, ct, ln);
  }
  for (int f = threadIdx.x; f < 2 * 2 * 32; f += blockDim.x) {
    int kc = f >> 6, ct = (f >> 5) & 1, ln = f & 31;
    sF1F[f] = make_B_frag(f1W, HH / 2, kc, ct, ln);
  }
  if (threadIdx.x < HH) sFcb[threadIdx.x] = fcb[threadIdx.x];
  if (threadIdx.x < HH / 2) { sF1b[threadIdx.x] = f1b[threadIdx.x]; sF2[threadIdx.x] = f2W[threadIdx.x]; }
  __syncthreads();

  int wave = threadIdx.x >> 5, lane = threadIdx.x & 31;
  int tile = blockIdx.x * 8 + wave;
  if (tile >= BB * NT) return;
  int b = tile / NT, t = tile % NT;
  int row0 = t * 16;
  int m = lane & 15, hh = lane >> 4, n = lane & 15;

  const float* xb = x + (size_t)b * NN * WW;
  const float* ab = agg + (size_t)b * NN * WW;

  // A tile: concat([x, agg]) rows row0..row0+15, K = 128 in 4 chunks
  v16h A[4];
  A[0] = load_A_g(xb + (size_t)row0 * WW, WW, m, hh, 0);
  A[1] = load_A_g(xb + (size_t)row0 * WW, WW, m, hh, 32);
  A[2] = load_A_g(ab + (size_t)row0 * WW, WW, m, hh, 0);
  A[3] = load_A_g(ab + (size_t)row0 * WW, WW, m, hh, 32);

  _Float16* myH = sH[wave];
#pragma unroll
  for (int ct = 0; ct < 4; ++ct) {
    v8f acc = {};
#pragma unroll
    for (int kc = 0; kc < 4; ++kc)
      acc = wmma_f16(A[kc], sFcF[(kc * 4 + ct) * 32 + lane], acc);
    float bias = sFcb[ct * 16 + n];
#pragma unroll
    for (int r = 0; r < 8; ++r) {
      float hv = acc[r] + bias;
      hv = hv > 0.f ? hv : 0.f;
      myH[(r + 8 * hh) * HH + ct * 16 + n] = (_Float16)hv;  // h tile, relu'd
    }
  }

  // layer 2: h[16x64] @ f1W[64x32]  (same-wave LDS ops stay in order)
  v16h hA0 = load_A_l(myH, HH, m, hh, 0);
  v16h hA1 = load_A_l(myH, HH, m, hh, 32);

  float* myMid = sMid[wave];
#pragma unroll
  for (int ct = 0; ct < 2; ++ct) {
    v8f acc = {};
    acc = wmma_f16(hA0, sF1F[(0 * 2 + ct) * 32 + lane], acc);
    acc = wmma_f16(hA1, sF1F[(1 * 2 + ct) * 32 + lane], acc);
    float bias = sF1b[ct * 16 + n];
    float w2   = sF2[ct * 16 + n];
#pragma unroll
    for (int r = 0; r < 8; ++r) {
      float mv = acc[r] + bias;
      mv = mv > 0.f ? mv : 0.f;
      myMid[(r + 8 * hh) * (HH / 2) + ct * 16 + n] = mv * w2;  // pre-multiplied
    }
  }

  // layer 3: per-row reduction of 32 products + f2b  (b128 LDS reads)
  if (lane < 16) {
    const v4f* rowp = (const v4f*)(myMid + lane * (HH / 2));
    float sum = f2b[0];
#pragma unroll
    for (int c = 0; c < (HH / 2) / 4; ++c) {
      v4f tq = rowp[c];
      sum += tq[0] + tq[1] + tq[2] + tq[3];
    }
    pred[(size_t)b * NN + row0 + lane] = sum;
  }
}

// ---------------------------------------------------------------- launcher
extern "C" void kernel_launch(void* const* d_in, const int* in_sizes, int n_in,
                              void* d_out, int out_size, void* d_ws, size_t ws_size,
                              hipStream_t stream) {
  const float* x   = (const float*)d_in[0];
  const int*   ei  = (const int*)  d_in[1];   // [2, E]
  const float* emb = (const float*)d_in[2];
  const float* Wq  = (const float*)d_in[3];
  const float* Wk  = (const float*)d_in[4];
  const float* v   = (const float*)d_in[5];
  const float* fcW = (const float*)d_in[6];
  const float* fcb = (const float*)d_in[7];
  const float* f1W = (const float*)d_in[8];
  const float* f1b = (const float*)d_in[9];
  const float* f2W = (const float*)d_in[10];
  const float* f2b = (const float*)d_in[11];
  float* pred = (float*)d_out;

  float* ws      = (float*)d_ws;
  float* q       = ws;                      // N*64
  float* kbuf    = q + (size_t)NN * DD;     // N*64
  float* score   = kbuf + (size_t)NN * DD;  // E (reused as exp)
  float* seg_max = score + EE;              // N
  float* seg_sum = seg_max + NN;            // N
  float* aggbuf  = seg_sum + NN;            // B*N*64

  gdn_init<<<(BB * NN * WW + 255) / 256, 256, 0, stream>>>(seg_max, seg_sum, aggbuf);

  gdn_qk<<<(NT + 7) / 8, 256, 0, stream>>>(emb, Wq, Wk, q, kbuf);

  gdn_score<<<EE / 8, 256, 0, stream>>>(ei, q, kbuf, v, score);

  gdn_segmax<<<(EE + 255) / 256, 256, 0, stream>>>(ei, score, seg_max);

  gdn_expsum<<<(EE + 255) / 256, 256, 0, stream>>>(ei, seg_max, score, seg_sum);

  gdn_agg<<<EE / 8, 256, 0, stream>>>(ei, x, score, seg_sum, aggbuf);

  gdn_mlp<<<(BB * NT + 7) / 8, 256, 0, stream>>>(x, aggbuf, fcW, fcb, f1W, f1b,
                                                 f2W, f2b, pred);
}
